// SelfAttention_43739946943227
// MI455X (gfx1250) — compile-verified, matrix operand order checked by reading, not run on hardware
//
#include <hip/hip_runtime.h>
#include <math.h>

// ---------------------------------------------------------------------------
// SAGAN-style self-attention, fused flash-attention formulation for gfx1250.
// B=8, H=W=64 (N=4096), C=256, F=64.
// Mixed precision: bf16 WMMA (v_wmma_f32_16x16x32_bf16), fp32 accumulate.
//
// Pipeline:
//   0) prep_weights : Wf/Wg/Wh/Wv  -> bf16, transposed [col][k] (K contiguous)
//   1) proj_kernel  : f = x*Wf, g = x*Wg (bf16 [N,64]); h = x*Wh stored
//                     transposed hvT[256][B*N] so V B-fragments are b128 loads
//   2) attn_kernel  : fused flash attention, S never hits memory
//   3) outproj      : out = gamma * (O @ Wv) + x
// ---------------------------------------------------------------------------

#define BATCH 8
#define NTOK  4096            // H*W
#define CDIM  256
#define FDIM  64
#define NROWS (BATCH * NTOK)  // 32768

typedef __bf16 v8bf  __attribute__((ext_vector_type(8)));
typedef __bf16 v16bf __attribute__((ext_vector_type(16)));
typedef float  v8f   __attribute__((ext_vector_type(8)));

__device__ __forceinline__ v16bf cat8(v8bf lo, v8bf hi) {
  return __builtin_shufflevector(lo, hi, 0,1,2,3,4,5,6,7,8,9,10,11,12,13,14,15);
}

__device__ __forceinline__ v8f wmma_bf16(v16bf a, v16bf b, v8f c) {
  // D = A(16x32) * B(32x16) + C, fp32 accumulate
  return __builtin_amdgcn_wmma_f32_16x16x32_bf16(false, a, false, b, (short)0, c,
                                                 false, false);
}

// 16x32 bf16 A/B fragment when the operand is stored with K contiguous
// (row stride `ld` applied by caller).  CDNA5 ISA 7.12.2 lane pattern:
// lanes 0-15 -> K = {0..7, 16..23}; lanes 16-31 -> K = {8..15, 24..31}.
// Two contiguous 16B loads per lane.
__device__ __forceinline__ v16bf load_frag(const __bf16* rowptr, int kb0) {
  v8bf lo = *(const v8bf*)(rowptr + kb0);
  v8bf hi = *(const v8bf*)(rowptr + kb0 + 16);
  return cat8(lo, hi);
}

// Build a fragment from two fp32 v8f chunks (pairs lower to v_cvt_pk_bf16_f32)
__device__ __forceinline__ v16bf frag_from_f32(v8f lo, v8f hi) {
  v16bf r;
  #pragma unroll
  for (int i = 0; i < 8; ++i) { r[i] = (__bf16)lo[i]; r[8 + i] = (__bf16)hi[i]; }
  return r;
}

// ---------------------------------------------------------------------------
// Kernel 0: one-time weight conversion, fp32 [256][cols] -> bf16 [cols][256]
// (transposed so K is contiguous: B-fragments become two b128 loads).
// ---------------------------------------------------------------------------
__device__ __forceinline__ void tconv(const float* __restrict__ W,
                                      __bf16* __restrict__ WT, int cols,
                                      int tid, int nthreads) {
  const int total = CDIM * cols;
  for (int i = tid; i < total; i += nthreads) {
    const int k = i / cols, c = i - k * cols;   // coalesced read of W
    WT[(size_t)c * CDIM + k] = (__bf16)W[i];
  }
}

__global__ __launch_bounds__(256) void prep_weights(
    const float* __restrict__ Wf, const float* __restrict__ Wg,
    const float* __restrict__ Wh, const float* __restrict__ Wv,
    __bf16* __restrict__ WfT, __bf16* __restrict__ WgT,
    __bf16* __restrict__ WhT, __bf16* __restrict__ WvT)
{
  const int tid = blockIdx.x * blockDim.x + threadIdx.x;
  const int nth = gridDim.x * blockDim.x;
  tconv(Wf, WfT, FDIM, tid, nth);
  tconv(Wg, WgT, FDIM, tid, nth);
  tconv(Wh, WhT, CDIM, tid, nth);
  tconv(Wv, WvT, CDIM, tid, nth);
}

// ---------------------------------------------------------------------------
// Kernel 1: projections.  One block = 16 rows of x, 8 waves; each wave
// computes 3 of the 24 16x16 output tiles, K = 256 in 8 WMMA steps.
// ---------------------------------------------------------------------------
__global__ __launch_bounds__(256) void proj_kernel(
    const float* __restrict__ x,
    const __bf16* __restrict__ WfT, const __bf16* __restrict__ WgT,
    const __bf16* __restrict__ WhT,
    __bf16* __restrict__ fq, __bf16* __restrict__ gk,
    __bf16* __restrict__ hvT)
{
  __shared__ __bf16 xs[16 * CDIM];   // 8 KB: 16 rows of x in bf16

  const int row0 = blockIdx.x * 16;       // global row in [0, NROWS)
  const int tid  = threadIdx.x;

  // stage x tile as bf16 via float4 loads
  for (int idx = tid * 4; idx < 16 * CDIM; idx += 256 * 4) {
    const float4 v = *(const float4*)&x[(size_t)row0 * CDIM + idx];
    xs[idx + 0] = (__bf16)v.x; xs[idx + 1] = (__bf16)v.y;
    xs[idx + 2] = (__bf16)v.z; xs[idx + 3] = (__bf16)v.w;
  }
  __syncthreads();

  const int lane = tid & 31;
  const int wave = tid >> 5;
  const int rc   = lane & 15;                 // A-row / B-col within tile
  const int kb0  = (lane < 16) ? 0 : 8;
  const int rb   = (lane < 16) ? 0 : 8;       // C/D row base for this lane half

  // A fragments shared by all 3 tiles of this wave (x tile, K=256)
  v16bf a[8];
  #pragma unroll
  for (int kb = 0; kb < 8; ++kb)
    a[kb] = load_frag(&xs[rc * CDIM + kb * 32], kb0);

  #pragma unroll
  for (int tt = 0; tt < 3; ++tt) {
    const int t = wave * 3 + tt;            // 0..23
    const __bf16* Wt; int colBase;
    if (t < 4)       { Wt = WfT; colBase = t * 16; }
    else if (t < 8)  { Wt = WgT; colBase = (t - 4) * 16; }
    else             { Wt = WhT; colBase = (t - 8) * 16; }

    // B fragments: WT[col][k], K contiguous -> two b128 loads each
    const __bf16* wp = Wt + (size_t)(colBase + rc) * CDIM;
    v8f acc = {};
    #pragma unroll
    for (int kb = 0; kb < 8; ++kb)
      acc = wmma_bf16(a[kb], load_frag(wp + kb * 32, kb0), acc);

    if (t < 8) {
      __bf16* dst = (t < 4) ? fq : gk;      // row-major [NROWS, 64]
      #pragma unroll
      for (int r = 0; r < 8; ++r)
        dst[(size_t)(row0 + rb + r) * FDIM + colBase + rc] = (__bf16)acc[r];
    } else {
      // transposed store: hvT[channel][global_row]
      #pragma unroll
      for (int r = 0; r < 8; ++r)
        hvT[(size_t)(colBase + rc) * NROWS + row0 + rb + r] = (__bf16)acc[r];
    }
  }
}

// ---------------------------------------------------------------------------
// Kernel 2: fused flash attention.  Each wave owns one 16-row tile and scans
// all 4096 keys in 32-column chunks:
//   S(16x32) = Q(16x64) * K^T     -> 4 WMMA
//   online softmax (shfl-xor row reductions, running m/l, rescale O)
//   O += P(16x32) * V(32x256)     -> 16 WMMA
// O accumulator: 16 x v8f = 128 VGPRs/lane.  P relayout via 2KB LDS/wave.
// ---------------------------------------------------------------------------
__global__ __launch_bounds__(256) void attn_kernel(
    const __bf16* __restrict__ fq, const __bf16* __restrict__ gk,
    const __bf16* __restrict__ hvT, float* __restrict__ obuf)
{
  __shared__ float pls[8][16 * 32];   // 16 KB total, per-wave P staging

  const int lane = threadIdx.x & 31;
  const int wave = threadIdx.x >> 5;
  const int tile = blockIdx.x * 8 + wave;   // 0..2047 row tiles
  const int b    = tile >> 8;               // 256 tiles per batch
  const int row0 = tile * 16;               // global row (b*NTOK + n0)

  const int rc  = lane & 15;
  const int kb0 = (lane < 16) ? 0 : 8;
  const int rb  = (lane < 16) ? 0 : 8;

  // Q fragments (K = 64 -> 2 fragments)
  const __bf16* qp = fq + (size_t)(row0 + rc) * FDIM;
  const v16bf aq0 = load_frag(qp, kb0);
  const v16bf aq1 = load_frag(qp + 32, kb0);

  v8f oacc[16];
  #pragma unroll
  for (int t = 0; t < 16; ++t) oacc[t] = (v8f){};

  float m_run[8], l_run[8];
  #pragma unroll
  for (int r = 0; r < 8; ++r) { m_run[r] = -3.0e38f; l_run[r] = 0.0f; }

  float* pb = pls[wave];

  for (int j = 0; j < NTOK / 32; ++j) {
    const int cb = j * 32;

    // ---- S = Q K^T for 32 columns (two 16x16 tiles) ----
    const __bf16* g0 = gk + (size_t)(b * NTOK + cb + rc) * FDIM;
    const __bf16* g1 = g0 + 16 * FDIM;
    v8f s0 = {}, s1 = {};
    s0 = wmma_bf16(aq0, load_frag(g0, kb0), s0);
    s0 = wmma_bf16(aq1, load_frag(g0 + 32, kb0), s0);
    s1 = wmma_bf16(aq0, load_frag(g1, kb0), s1);
    s1 = wmma_bf16(aq1, load_frag(g1 + 32, kb0), s1);

    if (j + 1 < NTOK / 32)  // prefetch next chunk of keys
      __builtin_prefetch(gk + (size_t)(b * NTOK + cb + 32 + rc) * FDIM, 0, 0);

    // ---- online softmax (no 1/sqrt(d) scaling, matching reference) ----
    float alpha[8];
    v8f p0, p1;
    #pragma unroll
    for (int r = 0; r < 8; ++r) {
      float mx = fmaxf(s0[r], s1[r]);
      #pragma unroll
      for (int off = 8; off >= 1; off >>= 1)
        mx = fmaxf(mx, __shfl_xor(mx, off, 32));   // reduce within 16-lane half
      const float mnew = fmaxf(m_run[r], mx);
      alpha[r] = __expf(m_run[r] - mnew);
      m_run[r] = mnew;
      p0[r] = __expf(s0[r] - mnew);
      p1[r] = __expf(s1[r] - mnew);
      float ls = p0[r] + p1[r];
      #pragma unroll
      for (int off = 8; off >= 1; off >>= 1)
        ls += __shfl_xor(ls, off, 32);
      l_run[r] = l_run[r] * alpha[r] + ls;
    }
    #pragma unroll
    for (int t = 0; t < 16; ++t)
      #pragma unroll
      for (int r = 0; r < 8; ++r)
        oacc[t][r] *= alpha[r];

    // ---- P: C/D layout -> A fragment layout via LDS ----
    #pragma unroll
    for (int r = 0; r < 8; ++r) {
      pb[(rb + r) * 32 + rc]      = p0[r];
      pb[(rb + r) * 32 + 16 + rc] = p1[r];
    }
    asm volatile("s_wait_dscnt 0" ::: "memory");
    const float* pr = pb + rc * 32 + kb0;
    const v16bf pa = frag_from_f32(*(const v8f*)pr, *(const v8f*)(pr + 16));

    // ---- O += P * V  (V from transposed hvT: contiguous along keys) ----
    const size_t vrow = (size_t)b * NTOK + cb;
    #pragma unroll
    for (int t = 0; t < 16; ++t) {
      const __bf16* vp = hvT + (size_t)(t * 16 + rc) * NROWS + vrow;
      oacc[t] = wmma_bf16(pa, load_frag(vp, kb0), oacc[t]);
    }
  }

  // ---- finalize: O /= l, store fp32 ----
  float linv[8];
  #pragma unroll
  for (int r = 0; r < 8; ++r) linv[r] = 1.0f / l_run[r];
  #pragma unroll
  for (int t = 0; t < 16; ++t) {
    #pragma unroll
    for (int r = 0; r < 8; ++r)
      obuf[(size_t)(row0 + rb + r) * CDIM + t * 16 + rc] = oacc[t][r] * linv[r];
  }
}

// ---------------------------------------------------------------------------
// Kernel 3: out = gamma * (O @ Wv) + x.   One block = 16 rows, 4 waves x 4
// channel tiles, K = 256 in 8 WMMA steps.
// ---------------------------------------------------------------------------
__global__ __launch_bounds__(128) void outproj_kernel(
    const float* __restrict__ obuf, const __bf16* __restrict__ WvT,
    const float* __restrict__ gamma, const float* __restrict__ x,
    float* __restrict__ out)
{
  const int row0 = blockIdx.x * 16;
  const int lane = threadIdx.x & 31;
  const int wave = threadIdx.x >> 5;
  const int rc   = lane & 15;
  const int kb0  = (lane < 16) ? 0 : 8;
  const int rb   = (lane < 16) ? 0 : 8;

  // A fragments: O rows, fp32 -> bf16 via 32B vector loads + pack converts
  const float* op = obuf + (size_t)(row0 + rc) * CDIM;
  v16bf a[8];
  #pragma unroll
  for (int kb = 0; kb < 8; ++kb) {
    const float* p = op + kb * 32 + kb0;
    a[kb] = frag_from_f32(*(const v8f*)p, *(const v8f*)(p + 16));
  }

  const float gm = gamma[0];

  #pragma unroll
  for (int tt = 0; tt < 4; ++tt) {
    const int colBase = (wave * 4 + tt) * 16;
    const __bf16* wp = WvT + (size_t)(colBase + rc) * CDIM;
    v8f acc = {};
    #pragma unroll
    for (int kb = 0; kb < 8; ++kb)
      acc = wmma_bf16(a[kb], load_frag(wp + kb * 32, kb0), acc);
    #pragma unroll
    for (int r = 0; r < 8; ++r) {
      const size_t idx = (size_t)(row0 + rb + r) * CDIM + colBase + rc;
      out[idx] = gm * acc[r] + x[idx];
    }
  }
}

// ---------------------------------------------------------------------------
extern "C" void kernel_launch(void* const* d_in, const int* in_sizes, int n_in,
                              void* d_out, int out_size, void* d_ws, size_t ws_size,
                              hipStream_t stream) {
  const float* x     = (const float*)d_in[0];
  const float* Wf    = (const float*)d_in[1];
  const float* Wg    = (const float*)d_in[2];
  const float* Wh    = (const float*)d_in[3];
  const float* Wv    = (const float*)d_in[4];
  const float* gamma = (const float*)d_in[5];
  float* out = (float*)d_out;

  char* ws = (char*)d_ws;
  __bf16* fq   = (__bf16*)(ws);                          //  4 MB  [NROWS,64]
  __bf16* gk   = (__bf16*)(ws + (4u  << 20));            //  4 MB  [NROWS,64]
  __bf16* hvT  = (__bf16*)(ws + (8u  << 20));            // 16 MB  [256,NROWS]
  float*  obuf = (float*) (ws + (24u << 20));            // 32 MB  [NROWS,256]
  char*   wb   = ws + (56u << 20);                       // 320 KB weights
  __bf16* WfT  = (__bf16*)(wb);                          // [64][256]
  __bf16* WgT  = (__bf16*)(wb + 32 * 1024);              // [64][256]
  __bf16* WhT  = (__bf16*)(wb + 64 * 1024);              // [256][256]
  __bf16* WvT  = (__bf16*)(wb + 192 * 1024);             // [256][256]

  prep_weights  <<<64, 256, 0, stream>>>(Wf, Wg, Wh, Wv, WfT, WgT, WhT, WvT);
  proj_kernel   <<<NROWS / 16, 256, 0, stream>>>(x, WfT, WgT, WhT, fq, gk, hvT);
  attn_kernel   <<<NROWS / 16 / 8, 256, 0, stream>>>(fq, gk, hvT, obuf);
  outproj_kernel<<<NROWS / 16, 128, 0, stream>>>(obuf, WvT, gamma, x, out);
}
